// Centroids_62852551409938
// MI455X (gfx1250) — matrix operand order repaired and checked by reading, not compiled, not run on hardware
//
#include <hip/hip_runtime.h>

// ---------------------------------------------------------------------------
// Centroid argmin for MI455X (gfx1250, wave32, WMMA + async global->LDS).
//
// score[i][j] = c2[j] - 2 * dot(latent[i], coords[j])   (x2[i] drops out)
// dot via bf16 hi/lo split: dot ~= hi.hi + hi.lo + lo.hi
//   (3x v_wmma_f32_16x16x32_bf16 per 32-wide k-chunk, K=128 -> 12 WMMA/tile)
//
// Pipeline: coords pre-converted to bf16 hi/lo in workspace (setup kernel),
// then streamed chunk-by-chunk into double-buffered LDS with
// GLOBAL_LOAD_ASYNC_TO_LDS_B128 (ASYNCcnt), overlapping copy with WMMA.
// ---------------------------------------------------------------------------

typedef __attribute__((ext_vector_type(16))) __bf16 v16bf;
typedef __attribute__((ext_vector_type(8)))  float  v8f;
typedef __attribute__((ext_vector_type(4)))  int    v4i;

struct U16x16 { unsigned short u[16]; };

static __device__ __forceinline__ unsigned short bf16_trunc_bits(float x) {
    return (unsigned short)(__builtin_bit_cast(unsigned int, x) >> 16);
}
static __device__ __forceinline__ float bf16_bits_to_f32(unsigned short b) {
    return __builtin_bit_cast(float, ((unsigned int)b) << 16);
}

#if __has_builtin(__builtin_amdgcn_global_load_async_to_lds_b128) && \
    __has_builtin(__builtin_amdgcn_s_wait_asynccnt)
#define USE_ASYNC 1
// Builtin signature (from clang diagnostic): (int4 AS1*, int4 AS3*, imm, imm).
// offset immediate is added to BOTH the global and LDS addresses (ISA 08 §4.4)
#define ASYNC_CP128(gsrc, ldst, off)                                          \
    __builtin_amdgcn_global_load_async_to_lds_b128(                           \
        (__attribute__((address_space(1))) v4i*)(gsrc),                       \
        (__attribute__((address_space(3))) v4i*)(ldst), (off), 0)
#define WAIT_ASYNC(n) __builtin_amdgcn_s_wait_asynccnt(n)
#else
#define USE_ASYNC 0
#define WAIT_ASYNC(n) ((void)0)
#endif

// ---- shared device helpers ------------------------------------------------

// Load one 16-row A tile k-chunk and split into bf16 hi/lo (A 16x32 layout:
// lane holds row M=lane&15; half h -> K = kc*32 + laneHi*16 + h).
static __device__ __forceinline__ void load_a_split(
        const float* __restrict__ arow, int kc, int laneHi,
        v16bf& a_hi, v16bf& a_lo) {
    U16x16 hb, lb;
#pragma unroll
    for (int g = 0; g < 2; ++g) {
        int k0 = kc * 32 + laneHi * 16 + g * 8;
        float4 v0 = *(const float4*)(arow + k0);
        float4 v1 = *(const float4*)(arow + k0 + 4);
        float vs[8] = {v0.x, v0.y, v0.z, v0.w, v1.x, v1.y, v1.z, v1.w};
#pragma unroll
        for (int e = 0; e < 8; ++e) {
            unsigned short h = bf16_trunc_bits(vs[e]);
            float r = vs[e] - bf16_bits_to_f32(h);
            hb.u[g * 8 + e] = h;
            lb.u[g * 8 + e] = bf16_trunc_bits(r);
        }
    }
    a_hi = __builtin_bit_cast(v16bf, hb);
    a_lo = __builtin_bit_cast(v16bf, lb);
}

// ---- kernel 1a: c2 only (fallback path) -----------------------------------
__global__ void centroids_c2_kernel(const float* __restrict__ coords,
                                    float* __restrict__ c2) {
    int j = blockIdx.x * blockDim.x + threadIdx.x;   // [0, 2048)
    const float4* row = (const float4*)(coords + (size_t)j * 128);
    float s = 0.f;
#pragma unroll
    for (int q = 0; q < 32; ++q) {
        float4 v = row[q];
        s += v.x * v.x + v.y * v.y + v.z * v.z + v.w * v.w;
    }
    c2[j] = s;
}

// ---- kernel 1b: c2 + pre-convert coords to bf16 hi/lo ---------------------
__global__ void centroids_setup_kernel(const float* __restrict__ coords,
                                       float* __restrict__ c2,
                                       unsigned int* __restrict__ chi_u,
                                       unsigned int* __restrict__ clo_u) {
    int j = blockIdx.x * blockDim.x + threadIdx.x;   // [0, 2048)
    const float4* row = (const float4*)(coords + (size_t)j * 128);
    float s = 0.f;
#pragma unroll
    for (int q = 0; q < 32; ++q) {
        float4 v = row[q];
        s += v.x * v.x + v.y * v.y + v.z * v.z + v.w * v.w;
        float vv[4] = {v.x, v.y, v.z, v.w};
        unsigned short h[4], l[4];
#pragma unroll
        for (int e = 0; e < 4; ++e) {
            h[e] = bf16_trunc_bits(vv[e]);
            float rr = vv[e] - bf16_bits_to_f32(h[e]);
            l[e] = bf16_trunc_bits(rr);
        }
        size_t base = (size_t)j * 64 + q * 2;        // uints (2 bf16 each)
        chi_u[base + 0] = (unsigned)h[0] | ((unsigned)h[1] << 16);
        chi_u[base + 1] = (unsigned)h[2] | ((unsigned)h[3] << 16);
        clo_u[base + 0] = (unsigned)l[0] | ((unsigned)l[1] << 16);
        clo_u[base + 1] = (unsigned)l[2] | ((unsigned)l[3] << 16);
    }
    c2[j] = s;
}

// ---- kernel 2 (fast): async-staged, double-buffered main loop -------------
// Grid: 512 blocks x 256 threads. Block covers 256 rows (8 waves x 2 tiles x 16).
__global__ __launch_bounds__(256) void centroids_argmin_fast(
        const float* __restrict__ latent,
        const unsigned short* __restrict__ chi,
        const unsigned short* __restrict__ clo,
        const float* __restrict__ c2,
        int* __restrict__ out) {
    __shared__ unsigned short lds_hi[2][64 * 128];   // 2 x 16 KB
    __shared__ unsigned short lds_lo[2][64 * 128];   // 2 x 16 KB

    const int tid    = threadIdx.x;
    const int lane   = tid & 31;
    const int wave   = tid >> 5;
    const int laneHi = lane >> 4;
    const int laneLo = lane & 15;
    const int rowBase = blockIdx.x * 256 + wave * 32;

    // A: 2 row-tiles x 4 k-chunks, bf16 hi/lo in VGPRs
    v16bf a_hi[2][4], a_lo[2][4];
#pragma unroll
    for (int rt = 0; rt < 2; ++rt) {
        const float* arow = latent + (size_t)(rowBase + rt * 16 + laneLo) * 128;
#pragma unroll
        for (int kc = 0; kc < 4; ++kc)
            load_a_split(arow, kc, laneHi, a_hi[rt][kc], a_lo[rt][kc]);
    }

    float bestV[2][8];
    int   bestI[2][8];
#pragma unroll
    for (int rt = 0; rt < 2; ++rt)
#pragma unroll
        for (int r = 0; r < 8; ++r) { bestV[rt][r] = 3.0e38f; bestI[rt][r] = 0; }

    // Per-chunk staging: 64 coords rows = 16 KB hi + 16 KB lo.
    // 256 threads -> 64 B (4 x b128) per thread per array.
    auto issue_stage = [&](int cc, int buf) {
        const unsigned short* gh = chi + (size_t)cc * 8192 + tid * 32;
        const unsigned short* gl = clo + (size_t)cc * 8192 + tid * 32;
        unsigned short* lh = &lds_hi[buf][tid * 32];
        unsigned short* ll = &lds_lo[buf][tid * 32];
#if USE_ASYNC
        ASYNC_CP128(gh, lh, 0);  ASYNC_CP128(gh, lh, 16);
        ASYNC_CP128(gh, lh, 32); ASYNC_CP128(gh, lh, 48);
        ASYNC_CP128(gl, ll, 0);  ASYNC_CP128(gl, ll, 16);
        ASYNC_CP128(gl, ll, 32); ASYNC_CP128(gl, ll, 48);
#else
        const uint4* sh = (const uint4*)gh; uint4* dh = (uint4*)lh;
        const uint4* sl = (const uint4*)gl; uint4* dl = (uint4*)ll;
#pragma unroll
        for (int q = 0; q < 4; ++q) { dh[q] = sh[q]; dl[q] = sl[q]; }
#endif
    };

    issue_stage(0, 0);
    for (int cc = 0; cc < 32; ++cc) {
        const int cur = cc & 1;
        if (cc + 1 < 32) issue_stage(cc + 1, cur ^ 1);   // prefetch next chunk
#if USE_ASYNC
        // async loads complete in order: <=8 outstanding => chunk cc landed
        if (cc + 1 < 32) WAIT_ASYNC(8); else WAIT_ASYNC(0);
#endif
        __syncthreads();

#pragma unroll
        for (int t = 0; t < 4; ++t) {
            const int colBase = cc * 64 + t * 16;
            const int lc = t * 16 + laneLo;

            v8f acc[2];
            acc[0] = v8f{0.f, 0.f, 0.f, 0.f, 0.f, 0.f, 0.f, 0.f};
            acc[1] = v8f{0.f, 0.f, 0.f, 0.f, 0.f, 0.f, 0.f, 0.f};

#pragma unroll
            for (int kc = 0; kc < 4; ++kc) {
                const unsigned short* ph = &lds_hi[cur][lc * 128 + kc * 32 + laneHi * 16];
                const unsigned short* pl = &lds_lo[cur][lc * 128 + kc * 32 + laneHi * 16];
                v16bf b_hi = *(const v16bf*)ph;
                v16bf b_lo = *(const v16bf*)pl;
#pragma unroll
                for (int rt = 0; rt < 2; ++rt) {
                    acc[rt] = __builtin_amdgcn_wmma_f32_16x16x32_bf16(
                        false, a_hi[rt][kc], false, b_hi, (short)0, acc[rt], false, false);
                    acc[rt] = __builtin_amdgcn_wmma_f32_16x16x32_bf16(
                        false, a_hi[rt][kc], false, b_lo, (short)0, acc[rt], false, false);
                    acc[rt] = __builtin_amdgcn_wmma_f32_16x16x32_bf16(
                        false, a_lo[rt][kc], false, b_hi, (short)0, acc[rt], false, false);
                }
            }

            const int   myCol = colBase + laneLo;
            const float c2v   = c2[myCol];
#pragma unroll
            for (int rt = 0; rt < 2; ++rt)
#pragma unroll
                for (int r = 0; r < 8; ++r) {
                    float s = c2v - 2.0f * acc[rt][r];
                    if (s < bestV[rt][r]) { bestV[rt][r] = s; bestI[rt][r] = myCol; }
                }
        }
        __syncthreads();
    }

#pragma unroll
    for (int rt = 0; rt < 2; ++rt)
#pragma unroll
        for (int r = 0; r < 8; ++r) {
            float v = bestV[rt][r];
            int   i = bestI[rt][r];
#pragma unroll
            for (int m = 1; m <= 8; m <<= 1) {
                float ov = __shfl_xor(v, m, 32);
                int   oi = __shfl_xor(i, m, 32);
                if (ov < v || (ov == v && oi < i)) { v = ov; i = oi; }
            }
            if (laneLo == 0)
                out[rowBase + rt * 16 + r + laneHi * 8] = i;
        }
}

// ---- kernel 2 (fallback, small ws): convert coords inside the loop --------
__global__ __launch_bounds__(256) void centroids_argmin_fallback(
        const float* __restrict__ latent,
        const float* __restrict__ coords,
        const float* __restrict__ c2,
        int* __restrict__ out) {
    __shared__ unsigned short lds_hi[64 * 128];
    __shared__ unsigned short lds_lo[64 * 128];

    const int tid    = threadIdx.x;
    const int lane   = tid & 31;
    const int wave   = tid >> 5;
    const int laneHi = lane >> 4;
    const int laneLo = lane & 15;
    const int rowBase = blockIdx.x * 256 + wave * 32;

    v16bf a_hi[2][4], a_lo[2][4];
#pragma unroll
    for (int rt = 0; rt < 2; ++rt) {
        const float* arow = latent + (size_t)(rowBase + rt * 16 + laneLo) * 128;
#pragma unroll
        for (int kc = 0; kc < 4; ++kc)
            load_a_split(arow, kc, laneHi, a_hi[rt][kc], a_lo[rt][kc]);
    }

    float bestV[2][8];
    int   bestI[2][8];
#pragma unroll
    for (int rt = 0; rt < 2; ++rt)
#pragma unroll
        for (int r = 0; r < 8; ++r) { bestV[rt][r] = 3.0e38f; bestI[rt][r] = 0; }

    for (int cc = 0; cc < 32; ++cc) {
        {
            int e0 = tid * 32;
            int r  = e0 >> 7;
            int c0 = e0 & 127;
            const float4* src = (const float4*)(coords + (size_t)(cc * 64 + r) * 128 + c0);
            unsigned int* dhi = (unsigned int*)lds_hi + ((r * 128 + c0) >> 1);
            unsigned int* dlo = (unsigned int*)lds_lo + ((r * 128 + c0) >> 1);
#pragma unroll
            for (int q = 0; q < 8; ++q) {
                float4 v = src[q];
                float vv[4] = {v.x, v.y, v.z, v.w};
                unsigned short h[4], l[4];
#pragma unroll
                for (int e = 0; e < 4; ++e) {
                    h[e] = bf16_trunc_bits(vv[e]);
                    float rr = vv[e] - bf16_bits_to_f32(h[e]);
                    l[e] = bf16_trunc_bits(rr);
                }
                dhi[q * 2 + 0] = (unsigned)h[0] | ((unsigned)h[1] << 16);
                dhi[q * 2 + 1] = (unsigned)h[2] | ((unsigned)h[3] << 16);
                dlo[q * 2 + 0] = (unsigned)l[0] | ((unsigned)l[1] << 16);
                dlo[q * 2 + 1] = (unsigned)l[2] | ((unsigned)l[3] << 16);
            }
        }
        __syncthreads();

#pragma unroll
        for (int t = 0; t < 4; ++t) {
            const int colBase = cc * 64 + t * 16;
            const int lc = t * 16 + laneLo;

            v8f acc[2];
            acc[0] = v8f{0.f, 0.f, 0.f, 0.f, 0.f, 0.f, 0.f, 0.f};
            acc[1] = v8f{0.f, 0.f, 0.f, 0.f, 0.f, 0.f, 0.f, 0.f};

#pragma unroll
            for (int kc = 0; kc < 4; ++kc) {
                const unsigned short* ph = lds_hi + lc * 128 + kc * 32 + laneHi * 16;
                const unsigned short* pl = lds_lo + lc * 128 + kc * 32 + laneHi * 16;
                v16bf b_hi = *(const v16bf*)ph;
                v16bf b_lo = *(const v16bf*)pl;
#pragma unroll
                for (int rt = 0; rt < 2; ++rt) {
                    acc[rt] = __builtin_amdgcn_wmma_f32_16x16x32_bf16(
                        false, a_hi[rt][kc], false, b_hi, (short)0, acc[rt], false, false);
                    acc[rt] = __builtin_amdgcn_wmma_f32_16x16x32_bf16(
                        false, a_hi[rt][kc], false, b_lo, (short)0, acc[rt], false, false);
                    acc[rt] = __builtin_amdgcn_wmma_f32_16x16x32_bf16(
                        false, a_lo[rt][kc], false, b_hi, (short)0, acc[rt], false, false);
                }
            }

            const int   myCol = colBase + laneLo;
            const float c2v   = c2[myCol];
#pragma unroll
            for (int rt = 0; rt < 2; ++rt)
#pragma unroll
                for (int r = 0; r < 8; ++r) {
                    float s = c2v - 2.0f * acc[rt][r];
                    if (s < bestV[rt][r]) { bestV[rt][r] = s; bestI[rt][r] = myCol; }
                }
        }
        __syncthreads();
    }

#pragma unroll
    for (int rt = 0; rt < 2; ++rt)
#pragma unroll
        for (int r = 0; r < 8; ++r) {
            float v = bestV[rt][r];
            int   i = bestI[rt][r];
#pragma unroll
            for (int m = 1; m <= 8; m <<= 1) {
                float ov = __shfl_xor(v, m, 32);
                int   oi = __shfl_xor(i, m, 32);
                if (ov < v || (ov == v && oi < i)) { v = ov; i = oi; }
            }
            if (laneLo == 0)
                out[rowBase + rt * 16 + r + laneHi * 8] = i;
        }
}

// ---------------------------------------------------------------------------
extern "C" void kernel_launch(void* const* d_in, const int* in_sizes, int n_in,
                              void* d_out, int out_size, void* d_ws, size_t ws_size,
                              hipStream_t stream) {
    (void)in_sizes; (void)n_in; (void)out_size;
    const float* latent = (const float*)d_in[0];   // [131072, 128] f32
    const float* coords = (const float*)d_in[1];   // [2048, 128]   f32
    int* out = (int*)d_out;                        // argmin indices (int32)

    // ws layout (fast path): c2[2048] f32 | chi[2048*128] bf16 | clo[2048*128] bf16
    const size_t C2_BYTES  = 2048 * sizeof(float);
    const size_t CVT_BYTES = (size_t)2048 * 128 * sizeof(unsigned short);
    const size_t REQ = C2_BYTES + 2 * CVT_BYTES;   // ~1.03 MB

    float* c2 = (float*)d_ws;

    if (ws_size >= REQ) {
        unsigned short* chi = (unsigned short*)((char*)d_ws + C2_BYTES);
        unsigned short* clo = chi + (size_t)2048 * 128;
        centroids_setup_kernel<<<2048 / 256, 256, 0, stream>>>(
            coords, c2, (unsigned int*)chi, (unsigned int*)clo);
        centroids_argmin_fast<<<131072 / 256, 256, 0, stream>>>(
            latent, chi, clo, c2, out);
    } else {
        centroids_c2_kernel<<<2048 / 256, 256, 0, stream>>>(coords, c2);
        centroids_argmin_fallback<<<131072 / 256, 256, 0, stream>>>(
            latent, coords, c2, out);
    }
}
